// TemporalEventConvolution_59906203844674
// MI455X (gfx1250) — matrix operand order repaired and checked by reading, not compiled, not run on hardware
//
#include <hip/hip_runtime.h>

#define B_ 8
#define N_ 4096
#define D_ 256
#define K_ 64
#define W_ 5

typedef __attribute__((ext_vector_type(16))) _Float16 v16h;
typedef __attribute__((ext_vector_type(8)))  _Float16 v8h;
typedef __attribute__((ext_vector_type(8)))  float    v8f;

__device__ __forceinline__ v16h combine16(v8h lo, v8h hi) {
  return __builtin_shufflevector(lo, hi, 0,1,2,3,4,5,6,7,8,9,10,11,12,13,14,15);
}

// A operand (16x32 f16): lanes 0-15 hold K 0-15, lanes 16-31 hold K 16-31
// (matches loadB). p -> row start of this 32-wide K chunk.
__device__ __forceinline__ v16h loadA(const _Float16* p, int lh) {
  v8h lo = *(const v8h*)(p + lh * 16);
  v8h hi = *(const v8h*)(p + lh * 16 + 8);
  return combine16(lo, hi);
}

// B operand (32x16 f16): lanes 0-15 hold K=0-15, lanes 16-31 hold K=16-31 of one
// output column; p -> K-major storage of this column's K chunk.
__device__ __forceinline__ v16h loadB(const _Float16* p, int lh) {
  return *(const v16h*)(p + lh * 16);
}

__device__ __forceinline__ v8f wmma16(v16h a, v16h b, v8f c) {
  return __builtin_amdgcn_wmma_f32_16x16x32_f16(false, a, false, b, (short)0, c,
                                                false, false);
}

// ---------------------------------------------------------------------------
// Kernel 1: stable argsort of time_idx per batch (bitonic, in LDS).
// key = (t << 12) | i  (t < 2^17, i < 2^12  ->  fits in 29 bits, stable ties)
// ---------------------------------------------------------------------------
__global__ __launch_bounds__(1024) void tec_sort(const int* __restrict__ tidx,
                                                 int* __restrict__ srt) {
  __shared__ unsigned int keys[N_];
  const int b = blockIdx.x;
  const int t = threadIdx.x;
  for (int i = t; i < N_; i += 1024)
    keys[i] = ((unsigned int)tidx[b * N_ + i] << 12) | (unsigned int)i;
  __syncthreads();
  for (int k = 2; k <= N_; k <<= 1) {
    for (int j = k >> 1; j > 0; j >>= 1) {
      for (int i = t; i < N_; i += 1024) {
        const int ij = i ^ j;
        if (ij > i) {
          const unsigned int a = keys[i], c = keys[ij];
          const bool up = ((i & k) == 0);
          if ((a > c) == up) { keys[i] = c; keys[ij] = a; }
        }
      }
      __syncthreads();
    }
  }
  for (int i = t; i < N_; i += 1024)
    srt[b * N_ + i] = (int)(keys[i] & 0xFFFu);
}

// ---------------------------------------------------------------------------
// Kernel 2: convert weights to f16 with WMMA-friendly layouts.
//   fF[k][j]   = filters[k][j]        (j = w*256+d, K-contig rows for B operand)
//   cT[d][k]   = centers[k][d]        (transposed -> K-contig per output col)
//   wT[d2][d]  = out_w[d2][d]         (rows already K-contig for y = ctx @ W^T)
// ---------------------------------------------------------------------------
__global__ __launch_bounds__(256) void tec_cvt(
    const float* __restrict__ F, const float* __restrict__ C,
    const float* __restrict__ Wm, _Float16* __restrict__ fF,
    _Float16* __restrict__ cT, _Float16* __restrict__ wT) {
  const int i = blockIdx.x * 256 + threadIdx.x;
  const int nF = K_ * W_ * D_;          // 81920
  const int nC = D_ * K_;               // 16384
  const int nW = D_ * D_;               // 65536
  if (i < nF) {
    fF[i] = (_Float16)F[i];
  } else if (i < nF + nC) {
    const int j = i - nF;
    const int d = j >> 6, k = j & 63;
    cT[j] = (_Float16)C[k * D_ + d];
  } else if (i < nF + nC + nW) {
    const int j = i - (nF + nC);
    wT[j] = (_Float16)Wm[j];
  }
}

// ---------------------------------------------------------------------------
// Kernel 3: fused window-conv -> softmax -> ctx -> out-proj -> layernorm ->
// mask^2 -> scatter (unsort). 8 waves/WG, one 16-row tile per wave.
// ---------------------------------------------------------------------------
__global__ __launch_bounds__(256) void tec_main(
    const float* __restrict__ q, const float* __restrict__ mask,
    const int* __restrict__ srt, const _Float16* __restrict__ fF,
    const _Float16* __restrict__ cT, const _Float16* __restrict__ wT,
    const float* __restrict__ ob, const float* __restrict__ ga,
    const float* __restrict__ be, float* __restrict__ out) {
  // q rows n0-2 .. n0+129 as f16, row stride 264 halfs (bank-conflict pad)
  __shared__ _Float16 qLds[132 * 264];
  // per-wave scratch: ctx f16 [16][264] (8448B, aliased by act f32 [16][68])
  //                 + probs f16 [16][72] (2304B)
  __shared__ __align__(16) unsigned char scrAll[8 * 10752];

  const int b  = blockIdx.y;
  const int n0 = blockIdx.x * 128;
  const int tid = threadIdx.x;
  const int wv = tid >> 5;
  const int ln = tid & 31;
  const int lr = ln & 15;   // row / column select inside tile
  const int lh = ln >> 4;   // half-wave select

  // ---- cooperative gather of sorted q rows into LDS (one row per iter) ----
  for (int r = 0; r < 132; ++r) {
    const int sr = n0 - 2 + r;
    float v = 0.0f;
    if (sr >= 0 && sr < N_) {
      const int g = srt[b * N_ + sr];
      v = q[((size_t)b * N_ + g) * D_ + tid];
    }
    qLds[r * 264 + tid] = (_Float16)v;
  }
  __syncthreads();

  _Float16* ctxLds = (_Float16*)(scrAll + wv * 10752);         // [16][264]
  float*    actLds = (float*)   (scrAll + wv * 10752);         // [16][68] alias
  _Float16* pLds   = (_Float16*)(scrAll + wv * 10752 + 8448);  // [16][72]

  const int rowBase = wv * 16;

  // ---- Stage A: logits[16][64] = Qwin[16][1280] x F^T ----
  // 40 K-chunks (c = w*8 + dc; filter offset = c*32). Ping-pong double
  // buffer: two named register sets alternate producer/consumer roles, so
  // loads land directly in the registers WMMA reads (no rotation copies).
  v8f acc[4];
#pragma unroll
  for (int kt = 0; kt < 4; ++kt)
#pragma unroll
    for (int e = 0; e < 8; ++e) acc[kt][e] = 0.0f;

  {
    const _Float16* fBase = fF + (size_t)lr * (W_ * D_);  // row k = kt*16+lr
    const int aBase = (rowBase + lr) * 264;
    const int FS = 16 * W_ * D_;  // filter k-tile stride (16 rows)

    v16h a0, a1, b0[4], b1[4];
    // preload chunk 0
    a0 = loadA(&qLds[aBase], lh);
#pragma unroll
    for (int kt = 0; kt < 4; ++kt) b0[kt] = loadB(fBase + kt * FS, lh);

    for (int i = 0; i < 19; ++i) {
      const int cA = 2 * i + 1;
      const int cB = 2 * i + 2;
      // load chunk cA into buffer 1
      a1 = loadA(&qLds[aBase + (cA >> 3) * 264 + (cA & 7) * 32], lh);
#pragma unroll
      for (int kt = 0; kt < 4; ++kt)
        b1[kt] = loadB(fBase + kt * FS + cA * 32, lh);
      // consume buffer 0 (chunk cB-2)
#pragma unroll
      for (int kt = 0; kt < 4; ++kt) acc[kt] = wmma16(a0, b0[kt], acc[kt]);
      // load chunk cB into buffer 0
      a0 = loadA(&qLds[aBase + (cB >> 3) * 264 + (cB & 7) * 32], lh);
#pragma unroll
      for (int kt = 0; kt < 4; ++kt)
        b0[kt] = loadB(fBase + kt * FS + cB * 32, lh);
      // consume buffer 1 (chunk cA)
#pragma unroll
      for (int kt = 0; kt < 4; ++kt) acc[kt] = wmma16(a1, b1[kt], acc[kt]);
    }
    // epilogue: chunk 38 in buffer 0, then chunk 39
    a1 = loadA(&qLds[aBase + 4 * 264 + 7 * 32], lh);
#pragma unroll
    for (int kt = 0; kt < 4; ++kt)
      b1[kt] = loadB(fBase + kt * FS + 39 * 32, lh);
#pragma unroll
    for (int kt = 0; kt < 4; ++kt) acc[kt] = wmma16(a0, b0[kt], acc[kt]);
#pragma unroll
    for (int kt = 0; kt < 4; ++kt) acc[kt] = wmma16(a1, b1[kt], acc[kt]);
  }

  // C-layout -> row-major f32 in LDS
#pragma unroll
  for (int kt = 0; kt < 4; ++kt)
#pragma unroll
    for (int g = 0; g < 8; ++g)
      actLds[(g + 8 * lh) * 68 + kt * 16 + lr] = acc[kt][g];
  __builtin_amdgcn_wave_barrier();

  // ---- softmax (scale 1/sqrt(D/4)=0.125); lanes 0..15 own one row each ----
  if (ln < 16) {
    float* arow = &actLds[ln * 68];
    float mx = -3.0e38f;
    for (int k = 0; k < K_; ++k) {
      const float t = arow[k] * 0.125f;
      arow[k] = t;
      mx = fmaxf(mx, t);
    }
    float s = 0.0f;
    for (int k = 0; k < K_; ++k) {
      const float e = __expf(arow[k] - mx);
      arow[k] = e;
      s += e;
    }
    const float inv = 1.0f / s;
    _Float16* prow = &pLds[ln * 72];
    for (int k = 0; k < K_; ++k) prow[k] = (_Float16)(arow[k] * inv);
  }
  __builtin_amdgcn_wave_barrier();

  // ---- Stage B: ctx[16][256] = P[16][64] x centers[64][256] -> f16 LDS ----
  for (int dt = 0; dt < 16; ++dt) {
    v8f c;
#pragma unroll
    for (int e = 0; e < 8; ++e) c[e] = 0.0f;
#pragma unroll
    for (int kc = 0; kc < 2; ++kc) {
      const v16h a  = loadA(&pLds[lr * 72 + kc * 32], lh);
      const v16h bm = loadB(cT + (size_t)(dt * 16 + lr) * K_ + kc * 32, lh);
      c = wmma16(a, bm, c);
    }
#pragma unroll
    for (int g = 0; g < 8; ++g)
      ctxLds[(g + 8 * lh) * 264 + dt * 16 + lr] = (_Float16)c[g];
  }
  __builtin_amdgcn_wave_barrier();

  // ---- Stage C: y = ctx x out_w^T + b; keep all tiles in registers ----
  v8f y[16];
  float s1[8], s2[8];
#pragma unroll
  for (int g = 0; g < 8; ++g) { s1[g] = 0.0f; s2[g] = 0.0f; }

#pragma unroll
  for (int dt = 0; dt < 16; ++dt) {
    v8f c;
#pragma unroll
    for (int e = 0; e < 8; ++e) c[e] = 0.0f;
#pragma unroll
    for (int kc = 0; kc < 8; ++kc) {
      const v16h a  = loadA(&ctxLds[lr * 264 + kc * 32], lh);
      const v16h bm = loadB(wT + (size_t)(dt * 16 + lr) * D_ + kc * 32, lh);
      c = wmma16(a, bm, c);
    }
    const float bias = ob[dt * 16 + lr];
#pragma unroll
    for (int g = 0; g < 8; ++g) {
      const float t = c[g] + bias;
      c[g] = t;
      s1[g] += t;
      s2[g] += t * t;
    }
    y[dt] = c;
  }

  // layernorm stats: reduce across the 16 lanes of each half-wave
#pragma unroll
  for (int off = 1; off < 16; off <<= 1) {
#pragma unroll
    for (int g = 0; g < 8; ++g) {
      s1[g] += __shfl_xor(s1[g], off);
      s2[g] += __shfl_xor(s2[g], off);
    }
  }

  float mean[8], rstd[8], mfac[8];
  int orow[8];
#pragma unroll
  for (int g = 0; g < 8; ++g) {
    mean[g] = s1[g] * (1.0f / 256.0f);
    const float var = s2[g] * (1.0f / 256.0f) - mean[g] * mean[g];
    rstd[g] = rsqrtf(var + 1e-5f);
    const int n = n0 + rowBase + g + 8 * lh;  // sorted-domain row
    const int o = srt[b * N_ + n];            // original position (scatter)
    orow[g] = o;
    const float m = mask[b * N_ + o];         // mask_sorted[n] == mask[o]
    mfac[g] = m * m;                          // applied twice in reference
  }

#pragma unroll
  for (int dt = 0; dt < 16; ++dt) {
    const int d2 = dt * 16 + lr;
    const float gg = ga[d2];
    const float bb = be[d2];
#pragma unroll
    for (int g = 0; g < 8; ++g) {
      const float t = (y[dt][g] - mean[g]) * rstd[g] * gg + bb;
      out[((size_t)b * N_ + orow[g]) * D_ + d2] = t * mfac[g];
    }
  }
}

// ---------------------------------------------------------------------------
extern "C" void kernel_launch(void* const* d_in, const int* in_sizes, int n_in,
                              void* d_out, int out_size, void* d_ws,
                              size_t ws_size, hipStream_t stream) {
  (void)in_sizes; (void)n_in; (void)out_size; (void)ws_size;
  const float* q    = (const float*)d_in[0];
  const int*   tidx = (const int*)d_in[1];
  const float* mask = (const float*)d_in[2];
  const float* F    = (const float*)d_in[3];
  const float* C    = (const float*)d_in[4];
  const float* Wm   = (const float*)d_in[5];
  const float* ob   = (const float*)d_in[6];
  const float* ga   = (const float*)d_in[7];
  const float* be   = (const float*)d_in[8];
  float* out = (float*)d_out;

  char* ws = (char*)d_ws;
  int*      srt = (int*)ws;                                    // 128 KB
  _Float16* fF  = (_Float16*)(ws + 131072);                    // 160 KB
  _Float16* cT  = (_Float16*)(ws + 131072 + 163840);           // 32 KB
  _Float16* wT  = (_Float16*)(ws + 131072 + 163840 + 32768);   // 128 KB

  tec_sort<<<B_, 1024, 0, stream>>>(tidx, srt);
  tec_cvt<<<640, 256, 0, stream>>>(F, C, Wm, fF, cT, wT);
  tec_main<<<dim3(N_ / 128, B_), 256, 0, stream>>>(q, mask, srt, fF, cT, wT,
                                                   ob, ga, be, out);
}
